// MultiHeadSelfAttention_83863531421698
// MI455X (gfx1250) — compile-verified, matrix operand order checked by reading, not run on hardware
//
#include <hip/hip_runtime.h>
#include <hip/hip_bf16.h>
#include <math.h>

#define D_MODEL 1024
#define SEQ     2048
#define BATCH   4
#define NHEADS  16
#define DHEAD   64
#define MTOT    (BATCH * SEQ)   // 8192 token rows

typedef __bf16 bf16_t;
typedef __attribute__((ext_vector_type(8)))  __bf16 v8bf;
typedef __attribute__((ext_vector_type(16))) __bf16 v16bf;
typedef __attribute__((ext_vector_type(8)))  float  v8f;

__device__ __forceinline__ v8f wmma_bf16(v16bf a, v16bf b, v8f c) {
  // D = A(16x32 bf16) x B(32x16 bf16) + C(16x16 f32)
  return __builtin_amdgcn_wmma_f32_16x16x32_bf16(
      /*neg_a=*/false, a, /*neg_b=*/false, b,
      /*c_mod=*/(short)0, c, /*reuse_a=*/false, /*reuse_b=*/false);
}

// Load one 16-bit WMMA operand fragment for this lane.
// A(16x32) and B(32x16) share the layout: lanes 0-15 hold k = {0..7,16..23},
// lanes 16-31 hold k = {8..15,24..31}. Caller passes p = row_base + klo
// (klo = 0 for lanes 0-15, 8 for lanes 16-31); we fetch p[0..7] and p[16..23].
__device__ __forceinline__ v16bf ld_frag(const bf16_t* p) {
  v8bf lo = *reinterpret_cast<const v8bf*>(p);        // 16B load
  v8bf hi = *reinterpret_cast<const v8bf*>(p + 16);   // 16B load
  v16bf r;
#pragma unroll
  for (int i = 0; i < 8; ++i) { r[i] = lo[i]; r[i + 8] = hi[i]; }
  return r;
}

// ---------------------------------------------------------------- fp32 -> bf16
__global__ void cvt_f32_bf16(const float* __restrict__ src,
                             bf16_t* __restrict__ dst, int n4) {
  int i = blockIdx.x * blockDim.x + threadIdx.x;
  if (i >= n4) return;
  const float4 v = reinterpret_cast<const float4*>(src)[i];
  union { bf16_t h[4]; uint2 u; } o;
  o.h[0] = (bf16_t)v.x; o.h[1] = (bf16_t)v.y;
  o.h[2] = (bf16_t)v.z; o.h[3] = (bf16_t)v.w;
  reinterpret_cast<uint2*>(dst)[i] = o.u;
}

// ------------------------------------------------------------- fused QKV GEMM
// y = x @ W^T + b for W in {Wq,Wk,Wv}. Each wave computes a 64(M)x64(N) tile:
// 4 M-subtiles reuse each B fragment -> 16 WMMA per 8 fragment loads.
// Q,K stored [B,H,S,Dh]; V stored transposed [B,H,Dh,S].
__global__ __launch_bounds__(256) void qkv_gemm(
    const bf16_t* __restrict__ xb,
    const bf16_t* __restrict__ wq, const bf16_t* __restrict__ wk,
    const bf16_t* __restrict__ wv,
    const float* __restrict__ bq, const float* __restrict__ bk,
    const float* __restrict__ bv,
    bf16_t* __restrict__ qb, bf16_t* __restrict__ kb,
    bf16_t* __restrict__ vtb) {
  const int wave = blockIdx.x * (blockDim.x >> 5) + (threadIdx.x >> 5);
  const int lane = threadIdx.x & 31;
  const int l16  = lane & 15;
  const int klo  = (lane >> 4) << 3;   // 0 or 8

  const int tile_n = wave % 48;        // 48*64 = 3072 output cols (Q|K|V)
  const int tile_m = wave / 48;        // 128 m-blocks of 64 rows
  const int m0 = tile_m * 64;
  const int n0 = tile_n * 64;
  const int which = n0 >> 10;          // 0:Q 1:K 2:V  (uniform per wave)
  const int nl0 = n0 & 1023;

  const bf16_t* w   = (which == 0) ? wq : (which == 1) ? wk : wv;
  const float*  bia = (which == 0) ? bq : (which == 1) ? bk : bv;

  const bf16_t* arow[4];
  const bf16_t* wrow[4];
  v8f c[4][4];                         // [m-subtile][n-subtile]
#pragma unroll
  for (int mt = 0; mt < 4; ++mt)
    arow[mt] = xb + (size_t)(m0 + mt * 16 + l16) * D_MODEL;
#pragma unroll
  for (int t = 0; t < 4; ++t) {
    wrow[t] = w + (size_t)(nl0 + t * 16 + l16) * D_MODEL;
    const float bini = bia[nl0 + t * 16 + l16];   // bias: same col, all rows
#pragma unroll
    for (int mt = 0; mt < 4; ++mt)
#pragma unroll
      for (int r = 0; r < 8; ++r) c[mt][t][r] = bini;
  }

  for (int k = 0; k < D_MODEL; k += 32) {
    v16bf a[4];
#pragma unroll
    for (int mt = 0; mt < 4; ++mt) {
      a[mt] = ld_frag(arow[mt] + k + klo);
      __builtin_prefetch(arow[mt] + k + 512, 0, 1);   // global_prefetch_b8
    }
#pragma unroll
    for (int t = 0; t < 4; ++t) {
      const v16bf b = ld_frag(wrow[t] + k + klo);
#pragma unroll
      for (int mt = 0; mt < 4; ++mt)
        c[mt][t] = wmma_bf16(a[mt], b, c[mt][t]);
    }
  }

  // D-layout: VGPR r -> row m0 + mt*16 + r + klo, col nl0 + t*16 + l16
#pragma unroll
  for (int t = 0; t < 4; ++t) {
    const int nl = nl0 + t * 16 + l16;
    const int h = nl >> 6, d = nl & 63;
#pragma unroll
    for (int mt = 0; mt < 4; ++mt) {
#pragma unroll
      for (int r = 0; r < 8; ++r) {
        const int m = m0 + mt * 16 + r + klo;
        const int b = m >> 11, s = m & (SEQ - 1);
        const bf16_t v = (bf16_t)c[mt][t][r];
        if (which == 0)
          qb[((size_t)(b * NHEADS + h) * SEQ + s) * DHEAD + d] = v;
        else if (which == 1)
          kb[((size_t)(b * NHEADS + h) * SEQ + s) * DHEAD + d] = v;
        else
          vtb[((size_t)(b * NHEADS + h) * DHEAD + d) * SEQ + s] = v;
      }
    }
  }
}

// ----------------------------------------------------------- flash attention
// 4 waves/WG; each wave owns 32 query rows (2 tiles) of one (b,h) and streams
// 32 keys/step, so each streamed K/V fragment feeds 2 query tiles.
__global__ __launch_bounds__(128) void flash_attn(
    const bf16_t* __restrict__ qb, const bf16_t* __restrict__ kb,
    const bf16_t* __restrict__ vtb, bf16_t* __restrict__ ab) {
  const int bh   = blockIdx.x >> 4;    // b*16 + h
  const int qblk = blockIdx.x & 15;    // SEQ/128 = 16 query blocks per head
  const int warp = threadIdx.x >> 5;
  const int lane = threadIdx.x & 31;
  const int l16  = lane & 15;
  const int klo  = (lane >> 4) << 3;

  const int q0 = qblk * 128 + warp * 32;

  __shared__ bf16_t Plds[4][2][16][32];   // per-wave, per-qt P tiles

  v16bf aq[2][2];                         // [query tile][dh half]
#pragma unroll
  for (int qt = 0; qt < 2; ++qt) {
    const bf16_t* qrow =
        qb + ((size_t)bh * SEQ + q0 + qt * 16 + l16) * DHEAD;
    aq[qt][0] = ld_frag(qrow + klo);        // dh 0..31
    aq[qt][1] = ld_frag(qrow + 32 + klo);   // dh 32..63
  }

  float mrow[2][8], lrow[2][8];
  v8f acc[2][4];
#pragma unroll
  for (int qt = 0; qt < 2; ++qt) {
#pragma unroll
    for (int r = 0; r < 8; ++r) {
      mrow[qt][r] = -__builtin_inff();
      lrow[qt][r] = 0.f;
    }
#pragma unroll
    for (int t = 0; t < 4; ++t)
#pragma unroll
      for (int r = 0; r < 8; ++r) acc[qt][t][r] = 0.f;
  }

  const float scale = 0.125f;          // 1/sqrt(64)

  for (int j0 = 0; j0 < SEQ; j0 += 32) {
    // Load streamed K fragments once, reuse for both query tiles.
    v16bf bk0[2], bk1[2];
#pragma unroll
    for (int kt = 0; kt < 2; ++kt) {
      const bf16_t* krow =
          kb + ((size_t)bh * SEQ + j0 + kt * 16 + l16) * DHEAD;
      bk0[kt] = ld_frag(krow + klo);
      bk1[kt] = ld_frag(krow + 32 + klo);
      __builtin_prefetch(krow + 64 * DHEAD, 0, 1);   // next-next key tile
    }

    // S = Q(32x64) @ K^T(64x32): 4 16x16 tiles, 2 chained WMMAs each (K=64)
    v8f st[2][2];
#pragma unroll
    for (int qt = 0; qt < 2; ++qt) {
#pragma unroll
      for (int kt = 0; kt < 2; ++kt) {
        v8f z;
#pragma unroll
        for (int r = 0; r < 8; ++r) z[r] = 0.f;
        z           = wmma_bf16(aq[qt][0], bk0[kt], z);
        st[qt][kt]  = wmma_bf16(aq[qt][1], bk1[kt], z);
      }
    }

    // Online softmax: row r of C sits in lanes 0-15 (row r) / 16-31 (row r+8);
    // xor-shuffles with masks 1..8 reduce exactly within each row.
#pragma unroll
    for (int qt = 0; qt < 2; ++qt) {
#pragma unroll
      for (int r = 0; r < 8; ++r) {
        float s0 = st[qt][0][r] * scale;
        float s1 = st[qt][1][r] * scale;
        float mx = fmaxf(s0, s1);
#pragma unroll
        for (int off = 1; off <= 8; off <<= 1)
          mx = fmaxf(mx, __shfl_xor(mx, off, 32));
        const float mnew = fmaxf(mrow[qt][r], mx);
        const float p0 = __expf(s0 - mnew);
        const float p1 = __expf(s1 - mnew);
        float ps = p0 + p1;
#pragma unroll
        for (int off = 1; off <= 8; off <<= 1)
          ps += __shfl_xor(ps, off, 32);
        const float corr = __expf(mrow[qt][r] - mnew);  // 0 on first step
        lrow[qt][r] = lrow[qt][r] * corr + ps;
        mrow[qt][r] = mnew;
        const int prow = r + klo;                       // D-layout row in tile
        Plds[warp][qt][prow][l16]      = (bf16_t)p0;
        Plds[warp][qt][prow][16 + l16] = (bf16_t)p1;
#pragma unroll
        for (int t = 0; t < 4; ++t) acc[qt][t][r] *= corr;
      }
    }

    // P(32x32) @ V(32x64). V fragments loaded once, reused by both q-tiles.
    // LDS round-trip converts P from D-layout to A-layout; same-wave LDS ops
    // are in-order on CDNA5 -> no barrier needed (per-wave private tiles).
    v16bf bvf[4];
#pragma unroll
    for (int t = 0; t < 4; ++t) {
      const bf16_t* vrow =
          vtb + ((size_t)bh * DHEAD + t * 16 + l16) * SEQ + j0;
      bvf[t] = ld_frag(vrow + klo);
    }
#pragma unroll
    for (int qt = 0; qt < 2; ++qt) {
      const v16bf ap = ld_frag(&Plds[warp][qt][l16][0] + klo);
#pragma unroll
      for (int t = 0; t < 4; ++t)
        acc[qt][t] = wmma_bf16(ap, bvf[t], acc[qt][t]);
    }
  }

  // Normalize and write attention output as [B*S, D] bf16 (head-concat cols)
  const int b = bh >> 4, h = bh & 15;
#pragma unroll
  for (int qt = 0; qt < 2; ++qt) {
#pragma unroll
    for (int r = 0; r < 8; ++r) {
      const float inv = 1.0f / lrow[qt][r];
      const int s = q0 + qt * 16 + r + klo;
#pragma unroll
      for (int t = 0; t < 4; ++t)
        ab[(size_t)(b * SEQ + s) * D_MODEL + h * DHEAD + t * 16 + l16] =
            (bf16_t)(acc[qt][t][r] * inv);
    }
  }
}

// --------------------------------------------------------- output projection
// Same 64x64-per-wave structure as qkv_gemm; fp32 result straight to d_out.
__global__ __launch_bounds__(256) void out_gemm(
    const bf16_t* __restrict__ ab, const bf16_t* __restrict__ wo,
    const float* __restrict__ bo, float* __restrict__ out) {
  const int wave = blockIdx.x * (blockDim.x >> 5) + (threadIdx.x >> 5);
  const int lane = threadIdx.x & 31;
  const int l16 = lane & 15, klo = (lane >> 4) << 3;
  const int tile_n = wave & 15;
  const int tile_m = wave >> 4;        // 128 m-blocks of 64 rows
  const int m0 = tile_m * 64, n0 = tile_n * 64;

  const bf16_t* arow[4];
  const bf16_t* wrow[4];
  v8f c[4][4];
#pragma unroll
  for (int mt = 0; mt < 4; ++mt)
    arow[mt] = ab + (size_t)(m0 + mt * 16 + l16) * D_MODEL;
#pragma unroll
  for (int t = 0; t < 4; ++t) {
    wrow[t] = wo + (size_t)(n0 + t * 16 + l16) * D_MODEL;
    const float bini = bo[n0 + t * 16 + l16];
#pragma unroll
    for (int mt = 0; mt < 4; ++mt)
#pragma unroll
      for (int r = 0; r < 8; ++r) c[mt][t][r] = bini;
  }

  for (int k = 0; k < D_MODEL; k += 32) {
    v16bf a[4];
#pragma unroll
    for (int mt = 0; mt < 4; ++mt) {
      a[mt] = ld_frag(arow[mt] + k + klo);
      __builtin_prefetch(arow[mt] + k + 512, 0, 1);
    }
#pragma unroll
    for (int t = 0; t < 4; ++t) {
      const v16bf b = ld_frag(wrow[t] + k + klo);
#pragma unroll
      for (int mt = 0; mt < 4; ++mt)
        c[mt][t] = wmma_bf16(a[mt], b, c[mt][t]);
    }
  }

#pragma unroll
  for (int mt = 0; mt < 4; ++mt)
#pragma unroll
    for (int t = 0; t < 4; ++t)
#pragma unroll
      for (int r = 0; r < 8; ++r)
        out[(size_t)(m0 + mt * 16 + r + klo) * D_MODEL + n0 + t * 16 + l16] =
            c[mt][t][r];
}

// ---------------------------------------------------------------------- host
extern "C" void kernel_launch(void* const* d_in, const int* in_sizes, int n_in,
                              void* d_out, int out_size, void* d_ws,
                              size_t ws_size, hipStream_t stream) {
  (void)in_sizes; (void)n_in; (void)out_size; (void)ws_size;
  const float* x  = (const float*)d_in[0];
  const float* Wq = (const float*)d_in[1];
  const float* bq = (const float*)d_in[2];
  const float* Wk = (const float*)d_in[3];
  const float* bk = (const float*)d_in[4];
  const float* Wv = (const float*)d_in[5];
  const float* bv = (const float*)d_in[6];
  const float* Wo = (const float*)d_in[7];
  const float* bo = (const float*)d_in[8];
  float* out = (float*)d_out;

  const size_t SZ_X = (size_t)MTOT * D_MODEL;      // 8,388,608 elements
  const size_t SZ_W = (size_t)D_MODEL * D_MODEL;   // 1,048,576 elements

  char* ws = (char*)d_ws;
  bf16_t* xb  = (bf16_t*)ws;  ws += SZ_X * 2;
  bf16_t* wqb = (bf16_t*)ws;  ws += SZ_W * 2;
  bf16_t* wkb = (bf16_t*)ws;  ws += SZ_W * 2;
  bf16_t* wvb = (bf16_t*)ws;  ws += SZ_W * 2;
  bf16_t* wob = (bf16_t*)ws;  ws += SZ_W * 2;
  bf16_t* qb  = (bf16_t*)ws;  ws += SZ_X * 2;      // [B,H,S,Dh]
  bf16_t* kb  = (bf16_t*)ws;  ws += SZ_X * 2;      // [B,H,S,Dh]
  bf16_t* vtb = (bf16_t*)ws;  ws += SZ_X * 2;      // [B,H,Dh,S] (transposed)
  bf16_t* ab  = (bf16_t*)ws;  ws += SZ_X * 2;      // attn out [B*S, D]

  cvt_f32_bf16<<<(int)((SZ_X / 4 + 255) / 256), 256, 0, stream>>>(x,  xb,  (int)(SZ_X / 4));
  cvt_f32_bf16<<<(int)((SZ_W / 4 + 255) / 256), 256, 0, stream>>>(Wq, wqb, (int)(SZ_W / 4));
  cvt_f32_bf16<<<(int)((SZ_W / 4 + 255) / 256), 256, 0, stream>>>(Wk, wkb, (int)(SZ_W / 4));
  cvt_f32_bf16<<<(int)((SZ_W / 4 + 255) / 256), 256, 0, stream>>>(Wv, wvb, (int)(SZ_W / 4));
  cvt_f32_bf16<<<(int)((SZ_W / 4 + 255) / 256), 256, 0, stream>>>(Wo, wob, (int)(SZ_W / 4));

  // 128 M-blocks * 48 N-tiles = 6144 waves / 8 waves-per-block
  qkv_gemm<<<768, 256, 0, stream>>>(xb, wqb, wkb, wvb, bq, bk, bv, qb, kb, vtb);
  // 64 heads * 16 query blocks of 128 rows, 4 waves each
  flash_attn<<<1024, 128, 0, stream>>>(qb, kb, vtb, ab);
  // 128 M-blocks * 16 N-tiles = 2048 waves / 8 waves-per-block
  out_gemm<<<256, 256, 0, stream>>>(ab, wob, bo, out);
}
